// mBP_model_91027536872111
// MI455X (gfx1250) — matrix operand order repaired and checked by reading, not compiled, not run on hardware
//
#include <hip/hip_runtime.h>

typedef __attribute__((ext_vector_type(16))) _Float16 v16h;
typedef __attribute__((ext_vector_type(8)))  _Float16 v8h_t;
typedef __attribute__((ext_vector_type(8)))  float    v8f;

#define NRAD   16
#define NCHAN  5      // NZETA + 1 radial channels
#define NL     35     // angular monomials for zmax=4
#define LPAD   48     // 35 padded to 3 WMMA N-tiles
#define FEAT   144    // NRAD * (1 + NLAMBDA*NZETA)

union H16 { v16h v; v8h_t p[2]; _Float16 h[16]; };
union F8  { v8f  v; float f[8]; };

__device__ __forceinline__ int lower_bound_i32(const int* __restrict__ a, int n, int key) {
  int lo = 0, hi = n;
  while (lo < hi) { int mid = (lo + hi) >> 1; if (a[mid] < key) lo = mid + 1; else hi = mid; }
  return lo;
}

__global__ __launch_bounds__(128)
void mbp_feats_kernel(const float* __restrict__ rij_unit,
                      const float* __restrict__ radial_ij,
                      const float* __restrict__ lambda_weights,
                      const float* __restrict__ fact_norm,
                      const int*   __restrict__ first_atom_idx,
                      const int*   __restrict__ lxlylz,
                      const int*   __restrict__ lxlylz_sum,
                      int nedge,
                      float* __restrict__ out)
{
  // k-contiguous (transposed) panels -> WMMA fragments are contiguous b128 LDS loads
  __shared__ alignas(64) _Float16 sAT[4][NRAD][32];   // [chan][rad j][edge k]
  __shared__ alignas(64) _Float16 sBT[LPAD][32];      // [monomial l][edge k]
  __shared__ alignas(16) float    sR4T[NRAD][32];     // [rad j][edge k] two-body channel
  __shared__ float    sG[4][NRAD][LPAD];              // accumulated g_i tiles
  __shared__ int      sLx[NL], sLy[NL], sLz[NL];
  __shared__ float    sFn[NL], sLam0[NL], sLam1[NL];

  const int t    = threadIdx.x;
  const int wave = t >> 5;       // zeta channel r = wave (z = r+1)
  const int lane = t & 31;
  const int atom = blockIdx.x;

  if (t < NL) {
    sLx[t] = lxlylz[3*t+0];
    sLy[t] = lxlylz[3*t+1];
    sLz[t] = lxlylz[3*t+2];
    sFn[t] = fact_norm[t];
    const float lw0 = lambda_weights[0];
    const float lw1 = lambda_weights[1];
    const int   s   = lxlylz_sum[t];
    float p0 = 1.f, p1 = 1.f;
    for (int q = 0; q < s; ++q) { p0 *= lw0; p1 *= lw1; }
    sLam0[t] = p0; sLam1[t] = p1;
  }
  // zero the padded monomial rows once
  for (int idx = t; idx < (LPAD - NL) * 32; idx += 128)
    sBT[NL + (idx >> 5)][idx & 31] = (_Float16)0.f;
  __syncthreads();

  const int start = lower_bound_i32(first_atom_idx, nedge, atom);
  const int end   = lower_bound_i32(first_atom_idx, nedge, atom + 1);
  const int cnt   = end - start;

  F8 acc0, acc1, acc2;
  #pragma unroll
  for (int i = 0; i < 8; ++i) { acc0.f[i] = 0.f; acc1.f[i] = 0.f; acc2.f[i] = 0.f; }
  float racc = 0.f;              // two-body segment sum (threads 0..15)

  for (int c0 = 0; c0 < cnt; c0 += 32) {
    // ---- cooperative, coalesced radial staging: 4 threads per edge ----
    {
      const int  ke    = t >> 2;              // edge slot 0..31
      const int  q     = t & 3;               // quarter of the 80-float record
      const bool val   = (c0 + ke) < cnt;
      const int  ec    = val ? (start + c0 + ke) : start;   // clamped: branch-free
      const float msk  = val ? 1.f : 0.f;
      const float4* rp = (const float4*)(radial_ij + (size_t)ec * (NRAD*NCHAN) + q*20);
      // prefetch next chunk's slice of this record stream
      __builtin_prefetch(radial_ij + ((size_t)ec + 32) * (NRAD*NCHAN) + q*20, 0, 0);
      float vv[20];
      #pragma unroll
      for (int w = 0; w < 5; ++w) {
        const float4 r = rp[w];
        vv[4*w+0] = r.x; vv[4*w+1] = r.y; vv[4*w+2] = r.z; vv[4*w+3] = r.w;
      }
      #pragma unroll
      for (int i = 0; i < 20; ++i) {
        const int j = (q << 2) + i / NCHAN;   // rad bin
        const int c = i % NCHAN;              // channel
        const float val_i = vv[i] * msk;
        if (c < 4) sAT[c][j][ke] = (_Float16)val_i;
        else       sR4T[j][ke]   = val_i;
      }
    }
    // ---- wave0: angular monomial panel ----
    if (wave == 0) {
      const bool val  = (c0 + lane) < cnt;
      const int  ec   = val ? (start + c0 + lane) : start;
      const float msk = val ? 1.f : 0.f;
      const float x = rij_unit[3*(size_t)ec + 0];
      const float y = rij_unit[3*(size_t)ec + 1];
      const float z = rij_unit[3*(size_t)ec + 2];
      float px[5], py[5], pz[5];
      px[0] = py[0] = pz[0] = 1.f;
      #pragma unroll
      for (int q = 1; q < 5; ++q) { px[q]=px[q-1]*x; py[q]=py[q-1]*y; pz[q]=pz[q-1]*z; }
      for (int l = 0; l < NL; ++l)
        sBT[l][lane] = (_Float16)(px[sLx[l]] * py[sLy[l]] * pz[sLz[l]] * sFn[l] * msk);
    }
    __syncthreads();

    // two-body reduction over the 32 edge slots (vectorized row reads)
    if (t < NRAD) {
      const float4* rr = (const float4*)&sR4T[t][0];
      float s = 0.f;
      #pragma unroll
      for (int w = 0; w < 8; ++w) { const float4 r = rr[w]; s += r.x + r.y + r.z + r.w; }
      racc += s;
    }

    // ---- fragments: contiguous-in-K b128 LDS loads ----
    H16 a;
    const int m   = lane & 15;
    const int kba = (lane >= 16) ? 16 : 0;     // lanes 0-15: K0-15; 16-31: K16-31
    a.p[0] = *(const v8h_t*)&sAT[wave][m][kba];
    a.p[1] = *(const v8h_t*)&sAT[wave][m][kba + 8];

    H16 b0, b1, b2;
    const int n   = lane & 15;
    const int kbb = (lane < 16) ? 0 : 16;      // lanes 0-15: K0-15; 16-31: K16-31
    b0.p[0] = *(const v8h_t*)&sBT[n     ][kbb];
    b0.p[1] = *(const v8h_t*)&sBT[n     ][kbb + 8];
    b1.p[0] = *(const v8h_t*)&sBT[n + 16][kbb];
    b1.p[1] = *(const v8h_t*)&sBT[n + 16][kbb + 8];
    b2.p[0] = *(const v8h_t*)&sBT[n + 32][kbb];
    b2.p[1] = *(const v8h_t*)&sBT[n + 32][kbb + 8];

    acc0.v = __builtin_amdgcn_wmma_f32_16x16x32_f16(false, a.v, false, b0.v, (short)0, acc0.v, false, false);
    acc1.v = __builtin_amdgcn_wmma_f32_16x16x32_f16(false, a.v, false, b1.v, (short)0, acc1.v, false, false);
    acc2.v = __builtin_amdgcn_wmma_f32_16x16x32_f16(false, a.v, false, b2.v, (short)0, acc2.v, false, false);
    __syncthreads();
  }

  // dump D tiles: VGPR i -> M = i (+8 for lanes 16-31), N = lane&15 (+16*tile)
  {
    const int rbase = (lane >= 16) ? 8 : 0;
    const int n     = lane & 15;
    #pragma unroll
    for (int i = 0; i < 8; ++i) {
      sG[wave][rbase + i][n]      = acc0.f[i];
      sG[wave][rbase + i][n + 16] = acc1.f[i];
      sG[wave][rbase + i][n + 32] = acc2.f[i];
    }
  }
  __syncthreads();

  // lambda contraction: gi3[j,nl] = 2^(1-z) * sum_l g^2 * lambda_nl^s(l)
  if (t < 64) {
    const int r = t >> 4;
    const int j = t & 15;
    float s0 = 0.f, s1 = 0.f;
    for (int l = 0; l < NL; ++l) {
      const float g  = sG[r][j][l];
      const float g2 = g * g;
      s0 += g2 * sLam0[l];
      s1 += g2 * sLam1[l];
    }
    const float scale = 1.0f / (float)(1 << r);   // 2^(1-z), z = r+1
    out[(size_t)atom*FEAT + NRAD + r*(2*NRAD) + 2*j + 0] = s0 * scale;
    out[(size_t)atom*FEAT + NRAD + r*(2*NRAD) + 2*j + 1] = s1 * scale;
  }
  if (t < NRAD) out[(size_t)atom*FEAT + t] = racc;
}

extern "C" void kernel_launch(void* const* d_in, const int* in_sizes, int n_in,
                              void* d_out, int out_size, void* d_ws, size_t ws_size,
                              hipStream_t stream) {
  const float* rij_unit       = (const float*)d_in[0];
  const float* radial_ij      = (const float*)d_in[1];
  const float* lambda_weights = (const float*)d_in[2];
  const float* fact_norm      = (const float*)d_in[3];
  const int*   first_atom_idx = (const int*)d_in[4];
  const int*   lxlylz         = (const int*)d_in[5];
  const int*   lxlylz_sum     = (const int*)d_in[6];
  const int nedge = in_sizes[4];
  const int nat   = out_size / FEAT;   // 144 features per atom

  mbp_feats_kernel<<<nat, 128, 0, stream>>>(
      rij_unit, radial_ij, lambda_weights, fact_norm,
      first_atom_idx, lxlylz, lxlylz_sum, nedge, (float*)d_out);
}
